// NNBackflowSlater2nd_62414464745760
// MI455X (gfx1250) — matrix operand order repaired and checked by reading, not compiled, not run on hardware
//
#include <hip/hip_runtime.h>

typedef __attribute__((ext_vector_type(16))) __bf16 v16bf;
typedef __attribute__((ext_vector_type(8)))  __bf16 v8bf;
typedef __attribute__((ext_vector_type(8)))  float  v8f;

__device__ __forceinline__ unsigned short f32_to_bf16_bits(float f) {
  union { float f; unsigned u; } x; x.f = f;
  unsigned r = (x.u + 0x7FFFu + ((x.u >> 16) & 1u)) >> 16;  // round-nearest-even
  return (unsigned short)r;
}

// src is K x N f32; dst is N x K bf16 (transposed), dst-linear indexing.
__global__ void cvt_f32_bf16_T(const float* __restrict__ src,
                               unsigned short* __restrict__ dst, int K, int N) {
  int i = blockIdx.x * blockDim.x + threadIdx.x;
  if (i < K * N) {
    int nn = i / K, kk = i - nn * K;
    dst[i] = f32_to_bf16_bits(src[(size_t)kk * N + nn]);
  }
}

__global__ void cvt_i32_bf16(const int* __restrict__ src,
                             unsigned short* __restrict__ dst, int count) {
  int i = blockIdx.x * blockDim.x + threadIdx.x;
  if (i < count) dst[i] = src[i] ? (unsigned short)0x3F80 : (unsigned short)0;
}

__device__ __forceinline__ void async_b128(unsigned lds_byte_off,
                                           const unsigned short* gptr) {
  asm volatile("global_load_async_to_lds_b128 %0, %1, off"
               :
               : "v"(lds_byte_off), "v"((unsigned long long)(size_t)gptr)
               : "memory");
}

__device__ __forceinline__ v16bf cat16(v8bf lo, v8bf hi) {
  return __builtin_shufflevector(lo, hi, 0, 1, 2, 3, 4, 5, 6, 7, 8, 9, 10, 11,
                                 12, 13, 14, 15);
}

// C[M,N] = act(A[M,K] @ WT[N,K]^T + bias). A row-major bf16, WT pre-transposed
// bf16, f32 accumulate. Tile BM=128, BN=64, BK=32; 256 threads = 8 waves; wave
// w owns rows [w*16,w*16+16) x 64 cols -> 4 v_wmma_f32_16x16x32_bf16 per step,
// issued back-to-back after preloading all five fragments. Tiles are
// double-buffered in LDS, filled by global_load_async_to_lds_b128 (ASYNCcnt),
// so the next tile fill overlaps the current tile's WMMAs.
template <bool TANH, bool OUTBF16>
__global__ __launch_bounds__(256) void gemm_bias_act(
    const unsigned short* __restrict__ A, const unsigned short* __restrict__ WT,
    const float* __restrict__ bias, void* __restrict__ Cout, int M, int N,
    int K) {
  __shared__ __align__(16) unsigned short As[2][128][32];  // [buf][m][k] 16 KB
  __shared__ __align__(16) unsigned short Bs[2][64][32];   // [buf][n][k]  8 KB

  const int tid  = threadIdx.x;
  const int wv   = tid >> 5;
  const int lane = tid & 31;
  const int g    = lane >> 4;   // half-wave group
  const int ln16 = lane & 15;

  const int n0 = blockIdx.x * 64;
  const int m0 = blockIdx.y * 128;

  // Per-thread async-copy slots: A-tile = 32 B/thread (2 x b128, LDS-linear
  // offset tid*32 == row (tid/2), k-seg (tid&1)*16). B-tile = 16 B/thread.
  const int aRow = tid >> 1, aSeg = (tid & 1) * 16;
  const int bRow = tid >> 2, bSeg = (tid & 3) * 8;

  const unsigned ldsA0 = (unsigned)(size_t)&As[0][0][0];
  const unsigned ldsA1 = (unsigned)(size_t)&As[1][0][0];
  const unsigned ldsB0 = (unsigned)(size_t)&Bs[0][0][0];
  const unsigned ldsB1 = (unsigned)(size_t)&Bs[1][0][0];

  // Persistent per-thread global pointers, advanced 32 elements per K-step.
  const unsigned short* ga = A + (size_t)(m0 + aRow) * K + aSeg;
  const unsigned short* gb = WT + (size_t)(n0 + bRow) * K + bSeg;

  auto issue_tile = [&](const unsigned short* pa, const unsigned short* pb,
                        int buf) {
    unsigned la = (buf ? ldsA1 : ldsA0) + (unsigned)tid * 32u;
    async_b128(la, pa);
    async_b128(la + 16u, pa + 8);
    unsigned lb = (buf ? ldsB1 : ldsB0) + (unsigned)tid * 16u;
    async_b128(lb, pb);
  };

  v8f acc0 = {}, acc1 = {}, acc2 = {}, acc3 = {};
  const int NK = K >> 5;

  issue_tile(ga, gb, 0);
  for (int kt = 0; kt < NK; ++kt) {
    const int cur = kt & 1;
    asm volatile("s_wait_asynccnt 0" ::: "memory");
    __syncthreads();  // buf[cur] complete in LDS for all waves
    if (kt + 1 < NK) issue_tile(ga + (kt + 1) * 32, gb + (kt + 1) * 32, cur ^ 1);

    // Preload ALL fragments into distinct registers, then burst the WMMAs.
    // A fragment: M = lane%16; K run = [g*16, +16)
    const int am = wv * 16 + ln16;
    v16bf af = cat16(*(const v8bf*)&As[cur][am][g * 16],
                     *(const v8bf*)&As[cur][am][g * 16 + 8]);
    const int kb = g * 16;  // B: N = lane%16, K run = [g*16, +16)
    v16bf bf0 = cat16(*(const v8bf*)&Bs[cur][ 0 + ln16][kb],
                      *(const v8bf*)&Bs[cur][ 0 + ln16][kb + 8]);
    v16bf bf1 = cat16(*(const v8bf*)&Bs[cur][16 + ln16][kb],
                      *(const v8bf*)&Bs[cur][16 + ln16][kb + 8]);
    v16bf bf2 = cat16(*(const v8bf*)&Bs[cur][32 + ln16][kb],
                      *(const v8bf*)&Bs[cur][32 + ln16][kb + 8]);
    v16bf bf3 = cat16(*(const v8bf*)&Bs[cur][48 + ln16][kb],
                      *(const v8bf*)&Bs[cur][48 + ln16][kb + 8]);

    acc0 = __builtin_amdgcn_wmma_f32_16x16x32_bf16(false, af, false, bf0,
                                                   (short)0, acc0, false, false);
    acc1 = __builtin_amdgcn_wmma_f32_16x16x32_bf16(false, af, false, bf1,
                                                   (short)0, acc1, false, false);
    acc2 = __builtin_amdgcn_wmma_f32_16x16x32_bf16(false, af, false, bf2,
                                                   (short)0, acc2, false, false);
    acc3 = __builtin_amdgcn_wmma_f32_16x16x32_bf16(false, af, false, bf3,
                                                   (short)0, acc3, false, false);
  }

  v8f accs[4] = {acc0, acc1, acc2, acc3};
#pragma unroll
  for (int t = 0; t < 4; ++t) {
    const float bb = bias[n0 + t * 16 + ln16];
#pragma unroll
    for (int v = 0; v < 8; ++v) {  // D: row = v + 8*(lane/16), col = lane%16
      int m = m0 + wv * 16 + g * 8 + v;
      float val = accs[t][v] + bb;
      if (TANH) val = tanhf(val);
      size_t i = (size_t)m * N + n0 + t * 16 + ln16;
      if (OUTBF16)
        ((unsigned short*)Cout)[i] = f32_to_bf16_bits(val);
      else
        ((float*)Cout)[i] = val;
    }
  }
}

// One block per sample; wave 0 = spin-up sector, wave 1 = spin-down.
// Gathers the 32 occupied rows of (M + F-block), LU with partial pivoting,
// emits log|det1*det2| + i*pi*(#negative signs) as interleaved complex64.
__global__ __launch_bounds__(64) void lu_logdet(
    const float* __restrict__ F,  // [CH, 8192] chunk of backflow outputs
    const int* __restrict__ n,    // [8192, 256] occupancies
    const float* __restrict__ M1, const float* __restrict__ M2,
    float* __restrict__ out, int bBase) {
  const int bLocal = blockIdx.x;
  const int b = bBase + bLocal;
  __shared__ int occ[2][128];
  __shared__ float A[2][32][33];
  __shared__ float res[2][2];

  const int tid = threadIdx.x;
  for (int i = tid; i < 256; i += 64) occ[i >> 7][i & 127] = n[(size_t)b * 256 + i];
  __syncthreads();

  const int w = tid >> 5, lane = tid & 31;

  // R = index of the (lane+1)-th occupied orbital (ascending, matches argsort)
  int R = 0;
  {
    int cnt = 0;
    for (int r = 0; r < 128; ++r)
      if (occ[w][r]) { if (cnt == lane) R = r; ++cnt; }
  }

  const float* Msec = (w == 0) ? M1 : M2;
  const float4* Fr = (const float4*)(F + (size_t)bLocal * 8192 + w * 4096 + R * 32);
  const float4* Mr = (const float4*)(Msec + R * 32);
#pragma unroll
  for (int c4 = 0; c4 < 8; ++c4) {
    float4 f = Fr[c4], m = Mr[c4];
    A[w][lane][c4 * 4 + 0] = f.x + m.x;
    A[w][lane][c4 * 4 + 1] = f.y + m.y;
    A[w][lane][c4 * 4 + 2] = f.z + m.z;
    A[w][lane][c4 * 4 + 3] = f.w + m.w;
  }
  __syncthreads();

  float logabs = 0.0f;
  int parity = 0;
  for (int k = 0; k < 32; ++k) {
    float v = (lane >= k) ? fabsf(A[w][lane][k]) : -1.0f;
    int idx = lane;
    for (int off = 16; off > 0; off >>= 1) {
      float v2 = __shfl_xor(v, off, 32);
      int i2 = __shfl_xor(idx, off, 32);
      if (v2 > v || (v2 == v && i2 < idx)) { v = v2; idx = i2; }
    }
    const int p = idx;  // wave-uniform
    if (p != k) {       // swap rows k,p; lane = column
      float t = A[w][k][lane];
      A[w][k][lane] = A[w][p][lane];
      A[w][p][lane] = t;
      parity ^= 1;
    }
    __syncthreads();
    const float pivv = A[w][k][k];
    if (pivv < 0.0f) parity ^= 1;
    logabs += logf(fabsf(pivv));
    if (lane > k) {
      float f = A[w][lane][k] / pivv;
      for (int c = k + 1; c < 32; ++c) A[w][lane][c] -= f * A[w][k][c];
    }
    __syncthreads();
  }

  if (lane == 0) { res[w][0] = logabs; res[w][1] = (float)parity; }
  __syncthreads();
  if (tid == 0) {
    out[2 * (size_t)b + 0] = res[0][0] + res[1][0];
    out[2 * (size_t)b + 1] = 3.14159265358979323846f * (res[0][1] + res[1][1]);
  }
}

extern "C" void kernel_launch(void* const* d_in, const int* in_sizes, int n_in,
                              void* d_out, int out_size, void* d_ws,
                              size_t ws_size, hipStream_t stream) {
  const int*   n  = (const int*)d_in[0];    // (8192, 256)
  const float* W1 = (const float*)d_in[1];  // (256, 512)
  const float* b1 = (const float*)d_in[2];  // (512,)
  const float* W2 = (const float*)d_in[3];  // (512, 8192)
  const float* b2 = (const float*)d_in[4];  // (8192,)
  const float* M1 = (const float*)d_in[5];  // (128, 32)
  const float* M2 = (const float*)d_in[6];  // (128, 32)
  float* out = (float*)d_out;               // complex64 interleaved (B,)

  const int B = 8192, NIN = 256, H = 512, NOUT = 8192;
  const int CH = 1024;  // F chunk: 1024 x 8192 f32 = 32 MiB (L2-resident)

  char* p = (char*)d_ws;
  unsigned short* Xbf  = (unsigned short*)p; p += (size_t)B * NIN * 2;    // 4 MiB
  unsigned short* W1T  = (unsigned short*)p; p += (size_t)NIN * H * 2;    // 256 KiB
  unsigned short* Hbf  = (unsigned short*)p; p += (size_t)B * H * 2;      // 8 MiB
  unsigned short* W2T  = (unsigned short*)p; p += (size_t)H * NOUT * 2;   // 8 MiB
  float* Fch = (float*)p;                                                 // 32 MiB

  cvt_f32_bf16_T<<<(NIN * H + 255) / 256, 256, 0, stream>>>(W1, W1T, NIN, H);
  cvt_f32_bf16_T<<<(H * NOUT + 255) / 256, 256, 0, stream>>>(W2, W2T, H, NOUT);
  cvt_i32_bf16<<<(B * NIN + 255) / 256, 256, 0, stream>>>(n, Xbf, B * NIN);

  // H = tanh(X @ W1 + b1), stored bf16 for the big GEMM
  gemm_bias_act<true, true><<<dim3(H / 64, B / 128), 256, 0, stream>>>(
      Xbf, W1T, b1, Hbf, B, H, NIN);

  // F = H @ W2 + b2 in chunks, each immediately consumed by the LU kernel
  for (int c0 = 0; c0 < B; c0 += CH) {
    gemm_bias_act<false, false><<<dim3(NOUT / 64, CH / 128), 256, 0, stream>>>(
        Hbf + (size_t)c0 * H, W2T, b2, Fch, CH, NOUT, H);
    lu_logdet<<<CH, 64, 0, stream>>>(Fch, n, M1, M2, out, c0);
  }
}
